// BasicNCA2D_91225105367407
// MI455X (gfx1250) — matrix operand order repaired and checked by reading, not compiled, not run on hardware
//
#include <hip/hip_runtime.h>
#include <hip/hip_bf16.h>

#define BSZ   4
#define HH    192
#define WW    192
#define CC    16
#define HID   128
#define HW    (HH*WW)          // 36864
#define NPIX  (BSZ*HW)         // 147456
#define NBLK  (NPIX/128)       // 1152 blocks of 128 pixels
#define NSTEP 10

typedef __attribute__((ext_vector_type(16))) __bf16 v16bf;
typedef __attribute__((ext_vector_type(8)))  float  v8f;
typedef __attribute__((ext_vector_type(4)))  int    v4i;

#if __has_builtin(__builtin_amdgcn_global_load_async_to_lds_b128) && \
    __has_builtin(__builtin_amdgcn_s_wait_asynccnt)
#define NCA_ASYNC 1
typedef __attribute__((address_space(1))) v4i* gv4i;   // global (AS1) v4i*
typedef __attribute__((address_space(3))) v4i* lv4i;   // LDS (AS3) v4i*
#endif

__device__ __forceinline__ float bfbits(unsigned int u16) {
    union { unsigned int i; float f; } v; v.i = u16 << 16; return v.f;
}

// sum across each 16-lane half via ds_swizzle xor-butterfly (SWAPX1/2/4/8)
#if __has_builtin(__builtin_amdgcn_ds_swizzle)
#define SWZ_ADD(v, pat) do { union { float f; int i; } _u; _u.f = (v); \
    _u.i = __builtin_amdgcn_ds_swizzle(_u.i, (pat)); (v) += _u.f; } while (0)
__device__ __forceinline__ float red16(float v) {
    SWZ_ADD(v, 0x041F);   // xor 1
    SWZ_ADD(v, 0x081F);   // xor 2
    SWZ_ADD(v, 0x101F);   // xor 4
    SWZ_ADD(v, 0x201F);   // xor 8
    return v;
}
#else
__device__ __forceinline__ float red16(float v) {
    for (int m = 1; m < 16; m <<= 1) v += __shfl_xor(v, m, 32);
    return v;
}
#endif

// ---------------- prep: NHWC -> NCHW state ----------------
__global__ void __launch_bounds__(256) nca_prep_x(const float* __restrict__ xin,
                                                  float* __restrict__ xs) {
    int i = blockIdx.x * 256 + threadIdx.x;
    if (i >= BSZ * CC * HW) return;
    int b = i / (CC * HW); int r = i - b * (CC * HW);
    int c = r / HW;        int yw = r - c * HW;
    xs[i] = xin[(size_t)(b * HW + yw) * CC + c];
}

// ---------------- prep: weights f32 -> bf16 ----------------
__global__ void __launch_bounds__(256) nca_prep_w(const float* __restrict__ wfc0f,
                                                  const float* __restrict__ wfc1f,
                                                  __bf16* __restrict__ wfc0,
                                                  __bf16* __restrict__ wfc1) {
    int i = blockIdx.x * 256 + threadIdx.x;
    if (i < HID * 32)                 wfc0[i] = (__bf16)wfc0f[i];
    else if (i < HID * 32 + CC * HID) wfc1[i - HID * 32] = (__bf16)wfc1f[i - HID * 32];
}

// ---------------- kernel 1: depthwise conv + fc0 (WMMA) + stat partials ----------------
// BN mean is recovered analytically from sum(dx) (fc0 is linear), so only the
// 32 dx-sums and 128 h^2-sums are reduced here.
__global__ void __launch_bounds__(256) nca_conv_fc0(
    const float*  __restrict__ xs,     // state NCHW
    const float*  __restrict__ w_p0,   // [16*81]
    const float*  __restrict__ b_p0,   // [16]
    const __bf16* __restrict__ wfc0,   // [128*32] bf16
    const float*  __restrict__ b_fc0,  // [128]
    __bf16*       __restrict__ hbuf,   // [NPIX*128] bf16, pixel-major
    float*        __restrict__ gpart)  // [NBLK*256]: [0:32) dx-sum, [128:256) h^2-sum
{
    __shared__ __bf16 dxs[8][16][32];          // per-wave dx tile [pixel][channel]
    __shared__ float  wdx[8][32];              // per-wave dx channel sums
    __shared__ float  wsq[8][128];             // per-wave h^2 channel sums

    int tid = threadIdx.x;
    int w = tid >> 5, l = tid & 31;
    int n = l & 15, chalf = l >> 4;
    int cbase = chalf * 8;
    int pbase = blockIdx.x * 128 + w * 16;
    int p = pbase + n;
    int b = p / HW; int rem = p - b * HW;
    int y = rem / WW; int x0 = rem - y * WW;

    // reflect tables (PAD=4, single reflection)
    int ry[9], rx[9];
#pragma unroll
    for (int t = 0; t < 9; ++t) {
        int yy = y - 4 + t;  yy = yy < 0 ? -yy : (yy >= HH ? 2 * HH - 2 - yy : yy);
        int xx = x0 - 4 + t; xx = xx < 0 ? -xx : (xx >= WW ? 2 * WW - 2 - xx : xx);
        ry[t] = yy; rx[t] = xx;
    }

    // each lane: copy 8 state channels + 8 depthwise-conv channels; reduce dx sums
#pragma unroll
    for (int j = 0; j < 8; ++j) {
        int ch = cbase + j;
        const float* xc = xs + (size_t)(b * CC + ch) * HW;
        float xv = xc[y * WW + x0];
        dxs[w][n][ch] = (__bf16)xv;
        float acc = b_p0[ch];
        const float* wp = w_p0 + ch * 81;
#pragma unroll
        for (int ty = 0; ty < 9; ++ty) {
            const float* xr = xc + ry[ty] * WW;
#pragma unroll
            for (int tx = 0; tx < 9; ++tx)
                acc = fmaf(wp[ty * 9 + tx], xr[rx[tx]], acc);
        }
        dxs[w][n][16 + ch] = (__bf16)acc;
        float rs = red16(xv);
        float rc = red16(acc);
        if (n == 0) { wdx[w][ch] = rs; wdx[w][16 + ch] = rc; }
    }
    __syncthreads();

    // B fragment: 32x16 bf16, element e <-> K = chalf*16 + e  (contiguous k order, matched by A)
    int koff = chalf * 16;
    v16bf bm;
#pragma unroll
    for (int e = 0; e < 16; ++e) bm[e] = dxs[w][n][koff + e];

    // 8 output tiles of 16 rows: h = Wfc0 * dx
#pragma unroll
    for (int t = 0; t < 8; ++t) {
        int m = t * 16 + n;
        v16bf am;
#pragma unroll
        for (int e = 0; e < 8; ++e) {          // A: elem e <-> K = khalf*16 + e
            am[e]     = wfc0[m * 32 + chalf * 16 + e];
            am[8 + e] = wfc0[m * 32 + chalf * 16 + 8 + e];
        }
        v8f c = {};
        c = __builtin_amdgcn_wmma_f32_16x16x32_bf16(false, am, false, bm,
                                                    (short)0, c, false, false);
        int mbase = t * 16 + chalf * 8;        // C: VGPR i -> M = mbase+i, N = n
        float hv[8];
#pragma unroll
        for (int i = 0; i < 8; ++i) hv[i] = c[i] + b_fc0[mbase + i];
#pragma unroll
        for (int i = 0; i < 8; ++i)
            hbuf[(size_t)p * HID + mbase + i] = (__bf16)hv[i];
#pragma unroll
        for (int i = 0; i < 8; ++i) {
            float q = red16(hv[i] * hv[i]);
            if (n == 0) wsq[w][mbase + i] = q;
        }
    }
    __syncthreads();
    if (tid < 32) {                            // fixed-order wave combine -> block partial
        float s = 0.f;
#pragma unroll
        for (int ww = 0; ww < 8; ++ww) s += wdx[ww][tid];
        gpart[(size_t)blockIdx.x * 256 + tid] = s;
    } else if (tid >= 128) {
        int cch = tid - 128;
        float q = 0.f;
#pragma unroll
        for (int ww = 0; ww < 8; ++ww) q += wsq[ww][cch];
        gpart[(size_t)blockIdx.x * 256 + tid] = q;
    }
}

// ---------------- deterministic cross-block reduce + analytic channel means ----------------
__global__ void __launch_bounds__(256) nca_reduce(const float* __restrict__ gpart,
                                                  const float* __restrict__ wfc0f,
                                                  const float* __restrict__ b_fc0,
                                                  float* __restrict__ gacc) {
    __shared__ float sdx[32];
    int tid = threadIdx.x;
    if (tid < 32) {
        float a = 0.f;
        for (int blk = 0; blk < NBLK; ++blk) a += gpart[(size_t)blk * 256 + tid];
        sdx[tid] = a;
    } else if (tid >= 128) {
        float a = 0.f;
        for (int blk = 0; blk < NBLK; ++blk) a += gpart[(size_t)blk * 256 + tid];
        gacc[tid] = a;                          // [128:256) = sum(h^2)
    }
    __syncthreads();
    if (tid < 128) {                            // sum(h) = Wfc0 * sum(dx) + N*b
        float s = b_fc0[tid] * (float)NPIX;
#pragma unroll
        for (int k = 0; k < 32; ++k) s = fmaf(wfc0f[tid * 32 + k], sdx[k], s);
        gacc[tid] = s;
    }
}

// ---------------- kernel 2: async tile -> LDS, BN + ReLU + fc1 (WMMA) + update ----------------
__global__ void __launch_bounds__(256) nca_fc1_update(
    const __bf16* __restrict__ hbuf,
    const __bf16* __restrict__ wfc1,   // [16*128] bf16
    const float*  __restrict__ gacc,   // [256] sums/sumsqs
    const float*  __restrict__ gamma_,
    const float*  __restrict__ beta_,
    const float*  __restrict__ ru,     // rand_u for this step, [NPIX]
    float*        __restrict__ xs)
{
    __shared__ float sscale[128], sshift[128];
#ifdef NCA_ASYNC
    __shared__ __bf16 hls[8][16][HID];         // per-wave 16px x 128ch tile (4 KB/wave)
#endif
    int tid = threadIdx.x;
    if (tid < 128) {
        const float invN = 1.0f / (float)NPIX;
        float mean = gacc[tid] * invN;
        float var  = gacc[128 + tid] * invN - mean * mean;
        float sc   = gamma_[tid] * rsqrtf(var + 1e-5f);
        sscale[tid] = sc;
        sshift[tid] = beta_[tid] - mean * sc;
    }
    __syncthreads();

    int w = tid >> 5, l = tid & 31, n = l & 15, chalf = l >> 4;
    int pbase = blockIdx.x * 128 + w * 16;
    int p = pbase + n;

#ifdef NCA_ASYNC
    {   // ASYNCcnt-tracked copy of this wave's h tile into LDS, 16B per lane-op
        const __bf16* gsrc = hbuf + (size_t)pbase * HID;
#pragma unroll
        for (int it = 0; it < 8; ++it) {
            int cidx = it * 32 + l;            // 256 chunks of 16B
            int px = cidx >> 4, c16 = cidx & 15;
            const __bf16* gp = gsrc + px * HID + c16 * 8;
            __builtin_amdgcn_global_load_async_to_lds_b128(
                (gv4i)(v4i*)(void*)const_cast<__bf16*>(gp),
                (lv4i)(v4i*)(void*)(&hls[w][px][c16 * 8]), 0, 0);
        }
        __builtin_amdgcn_s_wait_asynccnt(0);
    }
    const unsigned int* h32 = (const unsigned int*)(&hls[w][n][0]);
#else
    const unsigned int* h32 = (const unsigned int*)(hbuf + (size_t)p * HID);
#endif

    v8f c = {};
#pragma unroll
    for (int c4 = 0; c4 < 4; ++c4) {           // K = 128 as 4 chained WMMAs
        int kb = c4 * 32 + chalf * 16;
        v16bf bm;
#pragma unroll
        for (int v = 0; v < 8; ++v) {          // BN + ReLU fused into B assembly
            unsigned int u = h32[(kb >> 1) + v];
            int k0 = kb + 2 * v;
            float f0 = fmaxf(0.f, bfbits(u & 0xffffu) * sscale[k0]     + sshift[k0]);
            float f1 = fmaxf(0.f, bfbits(u >> 16)     * sscale[k0 + 1] + sshift[k0 + 1]);
            bm[2 * v]     = (__bf16)f0;
            bm[2 * v + 1] = (__bf16)f1;
        }
        v16bf am;
        int m = n;
#pragma unroll
        for (int e = 0; e < 8; ++e) {
            am[e]     = wfc1[m * HID + c4 * 32 + chalf * 16 + e];
            am[8 + e] = wfc1[m * HID + c4 * 32 + chalf * 16 + 8 + e];
        }
        c = __builtin_amdgcn_wmma_f32_16x16x32_bf16(false, am, false, bm,
                                                    (short)0, c, false, false);
    }

    // masked residual update, channel 0 frozen
    float mask = (ru[p] > 0.5f) ? 1.0f : 0.0f;
    int b = p / HW; int yw = p - b * HW;
    int mbase = chalf * 8;
#pragma unroll
    for (int i = 0; i < 8; ++i) {
        int m = mbase + i;
        if (m != 0) {
            size_t xi = (size_t)(b * CC + m) * HW + yw;
            xs[xi] = xs[xi] + c[i] * mask;
        }
    }
}

// ---------------- out: NCHW -> NHWC ----------------
__global__ void __launch_bounds__(256) nca_out(const float* __restrict__ xs,
                                               float* __restrict__ out) {
    int i = blockIdx.x * 256 + threadIdx.x;
    if (i >= BSZ * CC * HW) return;
    int c = i % CC; int t = i / CC; int yw = t % HW; int b = t / HW;
    out[i] = xs[(size_t)(b * CC + c) * HW + yw];
}

extern "C" void kernel_launch(void* const* d_in, const int* in_sizes, int n_in,
                              void* d_out, int out_size, void* d_ws, size_t ws_size,
                              hipStream_t stream) {
    (void)in_sizes; (void)n_in; (void)out_size; (void)ws_size;
    const float* x_in   = (const float*)d_in[0];
    const float* rand_u = (const float*)d_in[1];
    const float* w_p0   = (const float*)d_in[2];
    const float* b_p0   = (const float*)d_in[3];
    const float* w_fc0  = (const float*)d_in[4];
    const float* b_fc0  = (const float*)d_in[5];
    const float* w_fc1  = (const float*)d_in[6];
    const float* gamma_ = (const float*)d_in[7];
    const float* beta_  = (const float*)d_in[8];

    char* ws = (char*)d_ws;
    size_t off = 0;
    float*  xs    = (float*)(ws + off);  off += (size_t)BSZ * CC * HW * 4;      // 9.44 MB
    __bf16* hbuf  = (__bf16*)(ws + off); off += (size_t)NPIX * HID * 2;         // 37.75 MB
    __bf16* wfc0b = (__bf16*)(ws + off); off += HID * 32 * 2;                   // 8 KB
    __bf16* wfc1b = (__bf16*)(ws + off); off += CC * HID * 2;                   // 4 KB
    float*  gpart = (float*)(ws + off);  off += (size_t)NBLK * 256 * 4;         // 1.18 MB
    float*  gacc  = (float*)(ws + off);  off += 256 * 4;

    nca_prep_x<<<(BSZ * CC * HW + 255) / 256, 256, 0, stream>>>(x_in, xs);
    nca_prep_w<<<(HID * 32 + CC * HID + 255) / 256, 256, 0, stream>>>(w_fc0, w_fc1, wfc0b, wfc1b);

    for (int s = 0; s < NSTEP; ++s) {
        nca_conv_fc0<<<NBLK, 256, 0, stream>>>(xs, w_p0, b_p0, wfc0b, b_fc0, hbuf, gpart);
        nca_reduce<<<1, 256, 0, stream>>>(gpart, w_fc0, b_fc0, gacc);
        nca_fc1_update<<<NBLK, 256, 0, stream>>>(hbuf, wfc1b, gacc, gamma_, beta_,
                                                 rand_u + (size_t)s * NPIX, xs);
    }
    nca_out<<<(BSZ * CC * HW + 255) / 256, 256, 0, stream>>>(xs, (float*)d_out);
}
